// BasicQuantConv2d_47845935677802
// MI455X (gfx1250) — 8-bit matrix operand order repaired and checked by reading, compiled, not run on hardware
//
#include <hip/hip_runtime.h>
#include <hip/hip_bf16.h>

typedef int v8i __attribute__((ext_vector_type(8)));
typedef int v4i __attribute__((ext_vector_type(4)));

#define HWGQ_STEP  0.538f
#define HWGQ_THR   (3.0f * HWGQ_STEP)
#define GAUSS_STEP 0.996f
#define BN_EPS     1e-3f

#define C_IN   128
#define C_OUT  128
#define NIMG   32
#define HW     3136
#define KSTEPS 18
#define HW_TILE 64

#define WS_MEAN   0
#define WS_RSIG   512
#define WS_STATS  1024
#define WS_WPACK  4096
#define WS_AQ     151552

typedef __attribute__((address_space(1))) v4i as1_v4i;
typedef __attribute__((address_space(3))) v4i as3_v4i;

__device__ __forceinline__ void async_load16_to_lds(unsigned char* dst,
                                                    const unsigned char* src) {
#if __has_builtin(__builtin_amdgcn_global_load_async_to_lds_b128)
    __builtin_amdgcn_global_load_async_to_lds_b128(
        (as1_v4i*)src, (as3_v4i*)dst, 0, 0);
#else
    unsigned loff = (unsigned)(size_t)dst;
    asm volatile("global_load_async_to_lds_b128 %0, %1, off"
                 :: "v"(loff), "v"(src) : "memory");
#endif
}

__device__ __forceinline__ void wait_asynccnt0() {
#if __has_builtin(__builtin_amdgcn_s_wait_asynccnt)
    __builtin_amdgcn_s_wait_asynccnt(0);
#else
    asm volatile("s_wait_asynccnt 0x0" ::: "memory");
#endif
}

__global__ void bn_stats_kernel(const float* __restrict__ x,
                                float* __restrict__ meanOut,
                                float* __restrict__ rsigOut) {
    const int c = blockIdx.x;
    float s = 0.f, s2 = 0.f;
    for (int i = threadIdx.x; i < NIMG * HW; i += blockDim.x) {
        int n = i / HW;
        int hw = i - n * HW;
        float v = x[(size_t)(n * C_IN + c) * HW + hw];
        s += v; s2 += v * v;
    }
    __shared__ float ls[256], ls2[256];
    ls[threadIdx.x] = s; ls2[threadIdx.x] = s2;
    __syncthreads();
    for (int o = 128; o > 0; o >>= 1) {
        if ((int)threadIdx.x < o) {
            ls[threadIdx.x]  += ls[threadIdx.x + o];
            ls2[threadIdx.x] += ls2[threadIdx.x + o];
        }
        __syncthreads();
    }
    if (threadIdx.x == 0) {
        const float inv = 1.f / (float)(NIMG * HW);
        float m   = ls[0] * inv;
        float var = ls2[0] * inv - m * m;
        meanOut[c] = m;
        rsigOut[c] = rsqrtf(var + BN_EPS);
    }
}

__global__ void wstats_kernel(const float* __restrict__ w,
                              float* __restrict__ stats) {
    float s = 0.f, s2 = 0.f;
    for (int i = threadIdx.x; i < C_OUT * C_IN * 9; i += blockDim.x) {
        float v = w[i];
        s += v; s2 += v * v;
    }
    __shared__ float ls[256], ls2[256];
    ls[threadIdx.x] = s; ls2[threadIdx.x] = s2;
    __syncthreads();
    for (int o = 128; o > 0; o >>= 1) {
        if ((int)threadIdx.x < o) {
            ls[threadIdx.x]  += ls[threadIdx.x + o];
            ls2[threadIdx.x] += ls2[threadIdx.x + o];
        }
        __syncthreads();
    }
    if (threadIdx.x == 0) {
        const float inv = 1.f / (float)(C_OUT * C_IN * 9);
        float m   = ls[0] * inv;
        float var = fmaxf(ls2[0] * inv - m * m, 0.f);
        float step = GAUSS_STEP * sqrtf(var);
        stats[0] = step;
        stats[1] = HWGQ_STEP * 0.5f * step;
    }
}

__global__ void wpack_kernel(const float* __restrict__ w,
                             const float* __restrict__ stats,
                             signed char* __restrict__ wp) {
    int idx = blockIdx.x * blockDim.x + threadIdx.x;
    if (idx >= 8 * KSTEPS * 32 * 32) return;
    int b     = idx & 31;
    int lane  = (idx >> 5) & 31;
    int kstep = (idx >> 10) % KSTEPS;
    int mtile = idx / (KSTEPS * 1024);
    int v  = b >> 2;
    int bo = b & 3;
    int K  = ((v >> 2) << 5) + ((v & 3) << 2) + (((lane >> 4) & 1) << 4) + bo;
    int kk = kstep * 64 + K;
    int c  = kk & 127;
    int pos = kk >> 7;
    int r  = pos / 3;
    int s5 = pos - r * 3;
    int m  = mtile * 16 + (lane & 15);
    float wv = w[((size_t)(m * C_IN + c) * 3 + r) * 3 + s5];
    float step = stats[0];
    float inv  = (step > 0.f) ? (1.f / step) : 0.f;
    int code = 2 * (int)rintf(wv * inv + 0.5f) - 1;
    code = min(3, max(-3, code));
    wp[idx] = (signed char)code;
}

__global__ void aquant_kernel(const float* __restrict__ x,
                              const float* __restrict__ gamma,
                              const float* __restrict__ beta,
                              const float* __restrict__ mean,
                              const float* __restrict__ rsig,
                              unsigned char* __restrict__ aq) {
    __shared__ unsigned char tile[HW_TILE * C_IN];
    const int n   = blockIdx.y;
    const int hw0 = blockIdx.x * HW_TILE;
    const int hwl = threadIdx.x & (HW_TILE - 1);
    const int c0  = threadIdx.x >> 6;
    for (int c = c0; c < C_IN; c += 4) {
        float v  = x[(size_t)(n * C_IN + c) * HW + hw0 + hwl];
        float xn = (v - mean[c]) * rsig[c] * gamma[c] + beta[c];
        xn = fminf(fmaxf(xn, 0.f), HWGQ_THR);
        int code = (int)rintf(xn * (1.f / HWGQ_STEP));
        code = min(3, code);
        tile[hwl * C_IN + c] = (unsigned char)code;
    }
    __syncthreads();
    const unsigned int* t32 = (const unsigned int*)tile;
    unsigned int* o32 = (unsigned int*)(aq + ((size_t)n * HW + hw0) * C_IN);
    for (int i = threadIdx.x; i < HW_TILE * C_IN / 4; i += 256) o32[i] = t32[i];
}

__global__ __launch_bounds__(256, 1)
void qconv_wmma_kernel(const unsigned char* __restrict__ aq,
                       const signed char* __restrict__ wp,
                       const float* __restrict__ stats,
                       float* __restrict__ out) {
    __shared__ unsigned char lbsB[2][HW_TILE * 64];

    const int lane = threadIdx.x & 31;
    const int wave = threadIdx.x >> 5;
    const int n    = blockIdx.y;
    const int hw0  = blockIdx.x * HW_TILE;
    const int laneHalf = (lane >> 4) & 1;

    const unsigned char* aqn = aq + (size_t)n * HW * C_IN;

    const int sp   = threadIdx.x >> 2;
    const int sc16 = threadIdx.x & 3;
    int sph, spw;
    {
        int hw = hw0 + sp;
        sph = hw / 56;
        spw = hw - sph * 56;
    }

    v8i acc[4];
#pragma unroll
    for (int t = 0; t < 4; ++t)
#pragma unroll
        for (int j = 0; j < 8; ++j) acc[t][j] = 0;

    const signed char* wpw = wp + ((size_t)wave * KSTEPS) * 1024 + (size_t)lane * 32;

    auto stageB = [&](int kstep) {
        const int pos = kstep >> 1;
        const int r   = pos / 3;
        const int s5  = pos - r * 3;
        unsigned char* dst = &lbsB[kstep & 1][sp * 64 + sc16 * 16];
        int hs  = sph + r - 1;
        int ws2 = spw + s5 - 1;
        if (((unsigned)hs < 56u) && ((unsigned)ws2 < 56u)) {
            const unsigned char* src =
                aqn + (size_t)(hs * 56 + ws2) * C_IN + (kstep & 1) * 64 + sc16 * 16;
            async_load16_to_lds(dst, src);
        } else {
            v4i z; z[0] = 0; z[1] = 0; z[2] = 0; z[3] = 0;
            *(v4i*)dst = z;
        }
    };

    stageB(0);

    for (int kstep = 0; kstep < KSTEPS; ++kstep) {
        wait_asynccnt0();
        __syncthreads();

        if (kstep + 1 < KSTEPS)
            stageB(kstep + 1);

        const signed char* ap = wpw + (size_t)kstep * 1024;
        v4i a0 = *(const v4i*)(ap);
        v4i a1 = *(const v4i*)(ap + 16);
        v8i A  = __builtin_shufflevector(a0, a1, 0, 1, 2, 3, 4, 5, 6, 7);
        if (kstep + 1 < KSTEPS)
            __builtin_prefetch(ap + 1024, 0, 3);

        const unsigned char* bbase = lbsB[kstep & 1];
#pragma unroll
        for (int t = 0; t < 4; ++t) {
            const unsigned char* bp =
                bbase + (t * 16 + (lane & 15)) * 64 + laneHalf * 16;
            v4i b0 = *(const v4i*)(bp);
            v4i b1 = *(const v4i*)(bp + 32);
            v8i B  = __builtin_shufflevector(b0, b1, 0, 1, 2, 3, 4, 5, 6, 7);
            acc[t] = __builtin_amdgcn_wmma_i32_16x16x64_iu8(
                true, A, false, B, acc[t], false, false);
        }

        __syncthreads();
    }

    const float scale = stats[1];
    const int koff = wave * 16 + laneHalf * 8;
    const int hwst = hw0 + (lane & 15);
#pragma unroll
    for (int t = 0; t < 4; ++t)
#pragma unroll
        for (int j = 0; j < 8; ++j) {
            int k = koff + j;
            out[(size_t)(n * C_OUT + k) * HW + hwst + t * 16] =
                scale * (float)acc[t][j];
        }
}

extern "C" void kernel_launch(void* const* d_in, const int* in_sizes, int n_in,
                              void* d_out, int out_size, void* d_ws, size_t ws_size,
                              hipStream_t stream) {
    const float* x      = (const float*)d_in[0];
    const float* gamma  = (const float*)d_in[1];
    const float* beta   = (const float*)d_in[2];
    const float* weight = (const float*)d_in[3];
    float* out = (float*)d_out;

    char* ws = (char*)d_ws;
    float* mean  = (float*)(ws + WS_MEAN);
    float* rsig  = (float*)(ws + WS_RSIG);
    float* stats = (float*)(ws + WS_STATS);
    signed char*   wpack = (signed char*)(ws + WS_WPACK);
    unsigned char* aqbuf = (unsigned char*)(ws + WS_AQ);

    bn_stats_kernel<<<C_IN, 256, 0, stream>>>(x, mean, rsig);
    wstats_kernel<<<1, 256, 0, stream>>>(weight, stats);
    wpack_kernel<<<(8 * KSTEPS * 1024 + 255) / 256, 256, 0, stream>>>(weight, stats, wpack);
    aquant_kernel<<<dim3(HW / HW_TILE, NIMG), 256, 0, stream>>>(x, gamma, beta, mean, rsig, aqbuf);
    qconv_wmma_kernel<<<dim3(HW / HW_TILE, NIMG), 256, 0, stream>>>(aqbuf, wpack, stats, out);
}
